// LstmModel_54211077210485
// MI455X (gfx1250) — matrix operand order repaired and checked by reading, not compiled, not run on hardware
//
#include <hip/hip_runtime.h>

typedef __attribute__((ext_vector_type(16))) _Float16 v16h;
typedef __attribute__((ext_vector_type(8)))  float    v8f;
typedef __attribute__((ext_vector_type(4)))  unsigned int v4u;

#define Bn   512   // batch
#define Tn   256   // time steps
#define Hn   512   // hidden
#define NKT  16    // K tiles of 32 (512/32)
#define MT   64    // batch rows per workgroup

__device__ __forceinline__ float sigm(float x) { return 1.0f / (1.0f + __expf(-x)); }

// ---------------------------------------------------------------------------
// Zero the h ping-pong buffers + c state (first 2 MB of workspace).
// ---------------------------------------------------------------------------
__global__ void zero_ws_kernel(v4u* __restrict__ p) {
    int i = blockIdx.x * blockDim.x + threadIdx.x;   // exactly 131072 threads
    v4u z = {0u, 0u, 0u, 0u};
    p[i] = z;
}

// ---------------------------------------------------------------------------
// Pack W_hh [2048 x 512] fp32 row-major into f16 WMMA B-fragment order:
//   wpack[gt][kt][lane][i],  gt in [0,128) (16 gate-rows per tile),
//   kt in [0,16), lane in [0,32), i in [0,16)
// B order used here (contiguous k per lane half, not AMD table order): N = lane&15, K = kt*32 + (lane>=16 ? 16 : 0) + i
// ---------------------------------------------------------------------------
__global__ void pack_whh_kernel(const float* __restrict__ whh,
                                _Float16* __restrict__ wpack) {
    int idx  = blockIdx.x * blockDim.x + threadIdx.x;  // 2048*512 = 1,048,576
    int i    = idx & 15;
    int lane = (idx >> 4) & 31;
    int kt   = (idx >> 9) & 15;
    int gt   = idx >> 13;                              // 0..127
    int n    = gt * 16 + (lane & 15);                  // gate row (0..2047)
    int k    = kt * 32 + ((lane >> 4) * 16) + i;       // 0..511
    wpack[idx] = (_Float16)whh[n * Hn + k];
}

// ---------------------------------------------------------------------------
// One LSTM time step: gates = h_prev @ W_hh.T (+ x*W_ih + bias), cell update.
// Grid: (8 M-tiles, 32 hidden slices), block: 128 threads = 4 waves.
// Wave w computes gate w (i,f,g,o) for a [64 x 16] tile; fused cell epilogue.
// ---------------------------------------------------------------------------
__global__ __launch_bounds__(128)
void lstm_step_kernel(const _Float16* __restrict__ hin,
                      _Float16* __restrict__ hout,
                      float* __restrict__ cst,
                      const _Float16* __restrict__ wpack,
                      const float* __restrict__ seq,
                      const float* __restrict__ wih,
                      const float* __restrict__ bih,
                      const float* __restrict__ bhh,
                      int t) {
    // LDS union: A fragments (64 KB) then re-used as gate exchange buffer.
    __shared__ __align__(16) unsigned char smem[65536];
    _Float16* Apack = (_Float16*)smem;   // [4 ms][16 kt][32 lane][16 halfs]
    float*    gbuf  = (float*)smem;      // [4 gate][64 m][17] (padded)

    const int tid   = threadIdx.x;
    const int mtile = blockIdx.x;        // 0..7  (64 batch rows each)
    const int js    = blockIdx.y;        // 0..31 (16 hidden units each)

    // ---- Stage h tile [64 x 512] f16 into LDS, pre-swizzled to A-fragment
    // layout: lane<16 holds K {0..15}, lane>=16 holds K {16..31}.
    const _Float16* hbase = hin + (size_t)(mtile * MT) * Hn;
    #pragma unroll
    for (int j = 0; j < 32; ++j) {
        int chunk  = tid + 128 * j;          // 0..4095 chunks of 8 halfs
        int row    = chunk >> 6;             // 0..63
        int k8g    = chunk & 63;             // 8-half chunk within row
        int kt     = k8g >> 2;
        int sub    = k8g & 3;
        int laneHi = sub >> 1;
        int iHi    = sub & 1;
        int ms     = row >> 4;
        int lane_d = (row & 15) + (laneHi << 4);
        v4u data = *(const v4u*)(hbase + row * Hn + k8g * 8);
        *(v4u*)(Apack + (((ms * NKT + kt) * 32 + lane_d) * 16 + iHi * 8)) = data;
    }
    __syncthreads();

    // ---- WMMA main loop: wave w handles gate w over 4 M-subtiles.
    const int w    = tid >> 5;               // gate index 0..3
    const int lane = tid & 31;
    const int gt   = w * 32 + js;            // gate-row tile in wpack

    v8f acc[4] = {};
    for (int kt = 0; kt < NKT; ++kt) {
        v16h bfrag = *(const v16h*)(wpack +
            (((size_t)gt * NKT + kt) * 32 + lane) * 16);
        #pragma unroll
        for (int ms = 0; ms < 4; ++ms) {
            v16h afrag = *(const v16h*)(Apack +
                ((ms * NKT + kt) * 32 + lane) * 16);
            acc[ms] = __builtin_amdgcn_wmma_f32_16x16x32_f16(
                false, afrag, false, bfrag, (short)0, acc[ms], false, false);
        }
    }
    __syncthreads();   // all waves done reading Apack before gbuf overwrite

    // ---- Spill gate tiles to LDS for cross-wave cell update.
    // C/D layout: VGPR r, lane -> M = r + (lane>=16 ? 8 : 0), N = lane&15.
    const int hi = lane >> 4;
    const int ln = lane & 15;
    #pragma unroll
    for (int ms = 0; ms < 4; ++ms) {
        #pragma unroll
        for (int r = 0; r < 8; ++r) {
            int m = ms * 16 + hi * 8 + r;
            gbuf[(w * 64 + m) * 17 + ln] = acc[ms][r];
        }
    }
    __syncthreads();

    // ---- Fused LSTM cell update: 64x16 elements, 8 per thread.
    const int m0 = tid >> 1;                  // batch row within tile
    const int n0 = (tid & 1) * 8;             // hidden col start
    const int bg = mtile * MT + m0;           // global batch index
    const float x = seq[(size_t)bg * Tn + t]; // sequence[b, t, 0]
    const float cscale = 1.0f;
    (void)cscale;
    #pragma unroll
    for (int q = 0; q < 8; ++q) {
        int n   = n0 + q;
        int jgl = js * 16 + n;                // global hidden unit
        float gi = gbuf[(0 * 64 + m0) * 17 + n] + x * wih[jgl]
                 + bih[jgl] + bhh[jgl];
        float gf = gbuf[(1 * 64 + m0) * 17 + n] + x * wih[Hn + jgl]
                 + bih[Hn + jgl] + bhh[Hn + jgl];
        float gg = gbuf[(2 * 64 + m0) * 17 + n] + x * wih[2 * Hn + jgl]
                 + bih[2 * Hn + jgl] + bhh[2 * Hn + jgl];
        float go = gbuf[(3 * 64 + m0) * 17 + n] + x * wih[3 * Hn + jgl]
                 + bih[3 * Hn + jgl] + bhh[3 * Hn + jgl];
        size_t cidx = (size_t)bg * Hn + jgl;
        float cold = cst[cidx];
        float cnew = sigm(gf) * cold + sigm(gi) * tanhf(gg);
        float hnew = sigm(go) * tanhf(cnew);
        cst[cidx]  = cnew;
        hout[cidx] = (_Float16)hnew;
    }
}

// ---------------------------------------------------------------------------
// hid = relu(h @ fc1_w.T + fc1_b)    [512 x 256], K = 512
// ---------------------------------------------------------------------------
__global__ void fc1_kernel(const _Float16* __restrict__ h,
                           const float* __restrict__ w,
                           const float* __restrict__ b,
                           float* __restrict__ hid) {
    int idx = blockIdx.x * blockDim.x + threadIdx.x;  // 512*256
    int bb = idx >> 8;
    int u  = idx & 255;
    const _Float16* hr = h + (size_t)bb * Hn;
    const float*    wr = w + (size_t)u * Hn;
    float s = b[u];
    #pragma unroll 8
    for (int k = 0; k < Hn; ++k) s += (float)hr[k] * wr[k];
    hid[idx] = fmaxf(s, 0.0f);
}

// ---------------------------------------------------------------------------
// out = hid @ fc2_w.T + fc2_b        [512 x 28], K = 256
// ---------------------------------------------------------------------------
__global__ void fc2_kernel(const float* __restrict__ hid,
                           const float* __restrict__ w,
                           const float* __restrict__ b,
                           float* __restrict__ out) {
    int idx = blockIdx.x * blockDim.x + threadIdx.x;
    if (idx >= Bn * 28) return;
    int bb = idx / 28;
    int o  = idx % 28;
    const float* hr = hid + (size_t)bb * 256;
    const float* wr = w + (size_t)o * 256;
    float s = b[o];
    #pragma unroll 8
    for (int k = 0; k < 256; ++k) s += hr[k] * wr[k];
    out[idx] = s;
}

// ---------------------------------------------------------------------------
extern "C" void kernel_launch(void* const* d_in, const int* in_sizes, int n_in,
                              void* d_out, int out_size, void* d_ws, size_t ws_size,
                              hipStream_t stream) {
    (void)in_sizes; (void)n_in; (void)out_size; (void)ws_size;

    const float* seq = (const float*)d_in[0];  // [512,256,1]
    const float* wih = (const float*)d_in[1];  // [2048,1]
    const float* whh = (const float*)d_in[2];  // [2048,512]
    const float* bih = (const float*)d_in[3];  // [2048]
    const float* bhh = (const float*)d_in[4];  // [2048]
    const float* f1w = (const float*)d_in[5];  // [256,512]
    const float* f1b = (const float*)d_in[6];  // [256]
    const float* f2w = (const float*)d_in[7];  // [28,256]
    const float* f2b = (const float*)d_in[8];  // [28]

    char* ws = (char*)d_ws;
    _Float16* hA    = (_Float16*)(ws + 0);          // 512*512 f16 = 512 KB
    _Float16* hB    = (_Float16*)(ws + (512u << 10));
    float*    cst   = (float*)(ws + (1024u << 10)); // 512*512 f32 = 1 MB
    _Float16* wpack = (_Float16*)(ws + (2048u << 10)); // 2048*512 f16 = 2 MB
    float*    hid   = (float*)(ws + (4096u << 10)); // 512*256 f32

    // h0 = c0 = 0 (workspace is poisoned, must zero explicitly)
    zero_ws_kernel<<<512, 256, 0, stream>>>((v4u*)ws);      // first 2 MB
    pack_whh_kernel<<<4096, 256, 0, stream>>>(whh, wpack);

    _Float16* hin = hA;
    _Float16* hout = hB;
    for (int t = 0; t < Tn; ++t) {
        lstm_step_kernel<<<dim3(8, 32), 128, 0, stream>>>(
            hin, hout, cst, wpack, seq, wih, bih, bhh, t);
        _Float16* tmp = hin; hin = hout; hout = tmp;
    }

    fc1_kernel<<<512, 256, 0, stream>>>(hin, f1w, f1b, hid);
    fc2_kernel<<<(Bn * 28 + 255) / 256, 256, 0, stream>>>(hid, f2w, f2b,
                                                          (float*)d_out);
}
